// MuonLonghorn_61641370632604
// MI455X (gfx1250) — compile-verified, matrix operand order checked by reading, not run on hardware
//
#include <hip/hip_runtime.h>
#include <hip/hip_bf16.h>
#include <math.h>

typedef __attribute__((ext_vector_type(16))) __bf16 v16bf;
typedef __attribute__((ext_vector_type(8)))  __bf16 v8bf;
typedef __attribute__((ext_vector_type(8)))  float  v8f;

#define D_MODEL 768
#define D_INNER 1536
#define D_STATE 16
#define DT_RANK 48
#define XDBL_W  80            // DT_RANK + 2*D_STATE
#define SEQ_B   2
#define SEQ_L   1024
#define MTOT    (SEQ_B * SEQ_L)   // 2048 GEMM rows

// ---------------------------------------------------------------------------
// Pack fp32 (R x K, leading dim ld) -> bf16 (Rp x Kp), zero padded.
// Guarantees branch-free, 16B-vectorizable fragment loads in the GEMM.
// ---------------------------------------------------------------------------
__global__ void pack_bf16_kernel(const float* __restrict__ src, int ld,
                                 int R, int K, __bf16* __restrict__ dst,
                                 int Kp, int total) {
    int idx = blockIdx.x * blockDim.x + threadIdx.x;
    if (idx >= total) return;
    int k = idx % Kp, r = idx / Kp;
    float v = (r < R && k < K) ? src[(size_t)r * ld + k] : 0.0f;
    dst[idx] = (__bf16)v;
}

// ---------------------------------------------------------------------------
// CDNA5 16-bit A fragment: lane holds two contiguous 8-element K-runs at
// kbase = hi*8 and kbase+16  ->  two 16-byte loads, no branches.
// ---------------------------------------------------------------------------
__device__ __forceinline__ v16bf load_frag(const __bf16* __restrict__ p) {
    v8bf a = *(const v8bf*)p;
    v8bf b = *(const v8bf*)(p + 16);
    v16bf f;
#pragma unroll
    for (int i = 0; i < 8; ++i) { f[i] = a[i]; f[i + 8] = b[i]; }
    return f;
}

// C[M,N](f32) = Ap[M x Kp](bf16) * Wp[Npad x Kp](bf16)^T
// 4 waves / block tile M; each wave computes a 16(M) x 64(N) strip:
// per K-step: 1 A-frag + 4 B-frags + 4 v_wmma.
__global__ void __launch_bounds__(128)
wmma_gemm_bf16(const __bf16* __restrict__ Ap,
               const __bf16* __restrict__ Wp,
               float* __restrict__ C, int N, int Kp) {
    const int lane = threadIdx.x & 31;
    const int wave = threadIdx.x >> 5;
    const int hi = lane >> 4;
    const int lo = lane & 15;
    const int tileMbase = (blockIdx.y * 4 + wave) * 16;
    const int colbase   = blockIdx.x * 64;

    const __bf16* arow = Ap + (size_t)(tileMbase + lo) * Kp + hi * 8;
    const __bf16* wrow = Wp + (size_t)(colbase + lo) * Kp + hi * 8;
    const size_t wstep = (size_t)16 * Kp;

    v8f acc0 = {}, acc1 = {}, acc2 = {}, acc3 = {};
    for (int k0 = 0; k0 < Kp; k0 += 32) {
        v16bf a  = load_frag(arow + k0);
        v16bf b0 = load_frag(wrow + k0);
        v16bf b1 = load_frag(wrow + wstep + k0);
        v16bf b2 = load_frag(wrow + 2 * wstep + k0);
        v16bf b3 = load_frag(wrow + 3 * wstep + k0);
        acc0 = __builtin_amdgcn_wmma_f32_16x16x32_bf16(false, a, false, b0, (short)0, acc0, false, false);
        acc1 = __builtin_amdgcn_wmma_f32_16x16x32_bf16(false, a, false, b1, (short)0, acc1, false, false);
        acc2 = __builtin_amdgcn_wmma_f32_16x16x32_bf16(false, a, false, b2, (short)0, acc2, false, false);
        acc3 = __builtin_amdgcn_wmma_f32_16x16x32_bf16(false, a, false, b3, (short)0, acc3, false, false);
    }
    // fp32 C/D layout: VGPR r -> M = r (lanes 0-15) / r+8 (lanes 16-31)
#pragma unroll
    for (int r = 0; r < 8; ++r) {
        int m = tileMbase + hi * 8 + r;
        float vals[4] = {acc0[r], acc1[r], acc2[r], acc3[r]};
#pragma unroll
        for (int t = 0; t < 4; ++t) {
            int n = colbase + t * 16 + lo;
            if (n < N) C[(size_t)m * N + n] = vals[t];
        }
    }
}

// ---------------------------------------------------------------------------
// Causal depthwise conv (taps=4) + bias + SiLU; emits f32 (for scan) and
// bf16 (GEMM-ready A for x_proj).
// ---------------------------------------------------------------------------
__global__ void conv_silu_kernel(const float* __restrict__ xz,
                                 const float* __restrict__ cw,
                                 const float* __restrict__ cb,
                                 float* __restrict__ xc,
                                 __bf16* __restrict__ xcb, int total) {
    int idx = blockIdx.x * blockDim.x + threadIdx.x;
    if (idx >= total) return;
    int d  = idx % D_INNER;
    int bl = idx / D_INNER;          // b*L + l
    int l  = bl % SEQ_L;
    float acc = cb[d];
    const float* w = cw + d * 4;
#pragma unroll
    for (int t = 0; t < 4; ++t) {
        int ll = l - 3 + t;
        if (ll >= 0)
            acc += w[t] * xz[((size_t)(bl - l + ll)) * (2 * D_INNER) + d];
    }
    float s = acc / (1.0f + __expf(-acc));
    xc[idx]  = s;
    xcb[idx] = (__bf16)s;
}

// ---------------------------------------------------------------------------
// Sequential selective scan. One 1024-thread workgroup per batch; each thread
// holds 24 (d,n) state pairs (v,h) in registers: n = tid&15, d = (tid>>4)+64j.
// G = u k^T is rank-1, so 1 Newton-Schulz step collapses to the quintic
// (a s + b s^3 + c s^5)/n * u k^T with n = ||u||*||k||, s = n/(n+1e-7).
// ---------------------------------------------------------------------------
__global__ void __launch_bounds__(1024)
scan_kernel(const float* __restrict__ xc,    // (B,L,D_INNER)
            const float* __restrict__ dt,    // (B,L,D_INNER)
            const float* __restrict__ xdbl,  // (B,L,80): [48:64)=k [64:80)=q
            const float* __restrict__ dtb,   // dt_head_b
            const float* __restrict__ Dvec,  // D
            float* __restrict__ y) {         // (B,L,D_INNER)
    const int b     = blockIdx.x;
    const int tid   = threadIdx.x;
    const int n     = tid & 15;
    const int dbase = tid >> 4;              // 0..63
    const int lane  = tid & 31;
    const int wid   = tid >> 5;              // 0..31

    __shared__ float sk[D_STATE], sq[D_STATE];
    __shared__ float su[D_INNER], sdtv[D_INNER];
    __shared__ float sred[32];
    __shared__ float sg;

    float v[24], h[24];
#pragma unroll
    for (int j = 0; j < 24; ++j) { v[j] = 0.0f; h[j] = 0.0f; }

    for (int t = 0; t < SEQ_L; ++t) {
        const size_t base80 = ((size_t)(b * SEQ_L + t)) * XDBL_W;
        if (tid < D_STATE) {
            sk[tid] = xdbl[base80 + DT_RANK + tid];
            sq[tid] = xdbl[base80 + DT_RANK + D_STATE + tid];
        }
        __syncthreads();

        float ksq = 0.0f;
#pragma unroll
        for (int i = 0; i < D_STATE; ++i) ksq += sk[i] * sk[i];

        const size_t baseD = ((size_t)(b * SEQ_L + t)) * D_INNER;
        float usq = 0.0f;
        for (int d = tid; d < D_INNER; d += 1024) {
            float dv = dt[baseD + d] + dtb[d];
            dv = 1.0f / (1.0f + __expf(-dv));          // sigmoid
            dv = dv / (1.0f + dv * ksq);               // Longhorn normalizer
            float u = dv * xc[baseD + d];
            su[d] = u; sdtv[d] = dv;
            usq += u * u;
        }
#pragma unroll
        for (int o = 16; o > 0; o >>= 1) usq += __shfl_xor(usq, o, 32);
        if (lane == 0) sred[wid] = usq;
        __syncthreads();
        if (tid < 32) {
            float r = sred[tid];
#pragma unroll
            for (int o = 16; o > 0; o >>= 1) r += __shfl_xor(r, o, 32);
            if (tid == 0) {
                float nrm = sqrtf(r * ksq);
                float s  = nrm / (nrm + 1e-7f);
                float s2 = s * s;
                float p  = s * (3.4445f + s2 * (-4.7750f + 2.0315f * s2));
                sg = p / fmaxf(nrm, 1e-30f);
            }
        }
        __syncthreads();

        const float kn  = sk[n];
        const float qn  = sq[n];
        const float gk  = 1.0f /*ALPHA*/ * sg * kn;
        const float kn2 = kn * kn;
#pragma unroll
        for (int j = 0; j < 24; ++j) {
            int d = dbase + (j << 6);
            v[j] = 0.9f /*BETA*/ * v[j] + gk * su[d];
            h[j] = (1.0f - sdtv[d] * kn2) * h[j] + v[j];
            float part = h[j] * qn;
#pragma unroll
            for (int o = 8; o > 0; o >>= 1) part += __shfl_xor(part, o, 16);
            if (n == 0) y[baseD + d] = part + Dvec[d] * xc[baseD + d];
        }
        __syncthreads();
    }
}

// y_gated = y * silu(z); emits bf16 directly (GEMM-ready A for out_proj)
__global__ void gate_kernel(const float* __restrict__ y,
                            const float* __restrict__ xz,
                            __bf16* __restrict__ ygb, int total) {
    int idx = blockIdx.x * blockDim.x + threadIdx.x;
    if (idx >= total) return;
    int d  = idx % D_INNER;
    int bl = idx / D_INNER;
    float z = xz[(size_t)bl * (2 * D_INNER) + D_INNER + d];
    ygb[idx] = (__bf16)(y[idx] * (z / (1.0f + __expf(-z))));
}

// ---------------------------------------------------------------------------
extern "C" void kernel_launch(void* const* d_in, const int* in_sizes, int n_in,
                              void* d_out, int out_size, void* d_ws, size_t ws_size,
                              hipStream_t stream) {
    (void)in_sizes; (void)n_in; (void)out_size; (void)ws_size;
    const float* hs        = (const float*)d_in[0];
    const float* in_proj_w = (const float*)d_in[1];
    const float* conv_w    = (const float*)d_in[2];
    const float* conv_b    = (const float*)d_in[3];
    const float* x_proj_w  = (const float*)d_in[4];
    const float* dt_head_w = (const float*)d_in[5];
    const float* dt_head_b = (const float*)d_in[6];
    const float* out_proj_w= (const float*)d_in[7];
    const float* Dvec      = (const float*)d_in[8];
    float* out = (float*)d_out;

    // ---- workspace carve-out (256B aligned regions) ----
    char* wp = (char*)d_ws;
    auto alloc = [&](size_t bytes) -> char* {
        char* r = wp; wp += (bytes + 255) & ~(size_t)255; return r;
    };
    float*  xz     = (float*)alloc((size_t)MTOT * 2 * D_INNER * 4);
    float*  xc     = (float*)alloc((size_t)MTOT * D_INNER * 4);
    float*  xdbl   = (float*)alloc((size_t)MTOT * XDBL_W * 4);
    float*  dtf    = (float*)alloc((size_t)MTOT * D_INNER * 4);
    float*  ybuf   = (float*)alloc((size_t)MTOT * D_INNER * 4);
    __bf16* hs_b   = (__bf16*)alloc((size_t)MTOT * D_MODEL * 2);
    __bf16* inW_b  = (__bf16*)alloc((size_t)2 * D_INNER * D_MODEL * 2);
    __bf16* xc_b   = (__bf16*)alloc((size_t)MTOT * D_INNER * 2);
    __bf16* xpW_b  = (__bf16*)alloc((size_t)128 * D_INNER * 2);   // 80 -> 128 rows
    __bf16* xdbl_b = (__bf16*)alloc((size_t)MTOT * 64 * 2);       // K 48 -> 64
    __bf16* dtW_b  = (__bf16*)alloc((size_t)D_INNER * 64 * 2);
    __bf16* yg_b   = (__bf16*)alloc((size_t)MTOT * D_INNER * 2);
    __bf16* outW_b = (__bf16*)alloc((size_t)D_MODEL * D_INNER * 2);

    auto packN = [](int total) { return (total + 255) / 256; };

    // ---- pack inputs / weights to bf16 (padded) ----
    pack_bf16_kernel<<<packN(MTOT * D_MODEL), 256, 0, stream>>>(
        hs, D_MODEL, MTOT, D_MODEL, hs_b, D_MODEL, MTOT * D_MODEL);
    pack_bf16_kernel<<<packN(2 * D_INNER * D_MODEL), 256, 0, stream>>>(
        in_proj_w, D_MODEL, 2 * D_INNER, D_MODEL, inW_b, D_MODEL, 2 * D_INNER * D_MODEL);
    pack_bf16_kernel<<<packN(128 * D_INNER), 256, 0, stream>>>(
        x_proj_w, D_INNER, XDBL_W, D_INNER, xpW_b, D_INNER, 128 * D_INNER);
    pack_bf16_kernel<<<packN(D_INNER * 64), 256, 0, stream>>>(
        dt_head_w, DT_RANK, D_INNER, DT_RANK, dtW_b, 64, D_INNER * 64);
    pack_bf16_kernel<<<packN(D_MODEL * D_INNER), 256, 0, stream>>>(
        out_proj_w, D_INNER, D_MODEL, D_INNER, outW_b, D_INNER, D_MODEL * D_INNER);

    // 1) xz = hs @ in_proj_w^T            (2048 x 768 -> 3072)
    wmma_gemm_bf16<<<dim3((2 * D_INNER) / 64, MTOT / 64), 128, 0, stream>>>(
        hs_b, inW_b, xz, 2 * D_INNER, D_MODEL);

    // 2) depthwise causal conv + SiLU (f32 + bf16 outputs)
    conv_silu_kernel<<<packN(MTOT * D_INNER), 256, 0, stream>>>(
        xz, conv_w, conv_b, xc, xc_b, MTOT * D_INNER);

    // 3) x_dbl = xc @ x_proj_w^T          (2048 x 1536 -> 80, padded to 128)
    wmma_gemm_bf16<<<dim3(128 / 64, MTOT / 64), 128, 0, stream>>>(
        xc_b, xpW_b, xdbl, XDBL_W, D_INNER);

    // 4) dt = dt_low @ dt_head_w^T        (2048 x 48(pad 64) -> 1536)
    pack_bf16_kernel<<<packN(MTOT * 64), 256, 0, stream>>>(
        xdbl, XDBL_W, MTOT, DT_RANK, xdbl_b, 64, MTOT * 64);
    wmma_gemm_bf16<<<dim3(D_INNER / 64, MTOT / 64), 128, 0, stream>>>(
        xdbl_b, dtW_b, dtf, D_INNER, 64);

    // 5) sequential scan (rank-1-collapsed Newton-Schulz), one WG per batch
    scan_kernel<<<SEQ_B, 1024, 0, stream>>>(xc, dtf, xdbl, dt_head_b, Dvec, ybuf);

    // 6) gate with silu(z) -> bf16
    gate_kernel<<<packN(MTOT * D_INNER), 256, 0, stream>>>(
        ybuf, xz, yg_b, MTOT * D_INNER);

    // 7) out = yg @ out_proj_w^T          (2048 x 1536 -> 768)
    wmma_gemm_bf16<<<dim3(D_MODEL / 64, MTOT / 64), 128, 0, stream>>>(
        yg_b, outW_b, out, D_MODEL, D_INNER);
}
